// StylelizedTransformerBlock_5025111736929
// MI455X (gfx1250) — matrix operand order repaired and checked by reading, not compiled, not run on hardware
//
#include <hip/hip_runtime.h>

// ---------------- CDNA5 WMMA types / helpers -------------------------------
typedef __attribute__((ext_vector_type(16))) __bf16 v16bf;
typedef __attribute__((ext_vector_type(8)))  float  v8f;

#define NB 2
#define NT 1024
#define NC 256
#define NDH 8
#define NHD 64
#define NHN 4
#define NBH (NB*NDH)          // 16
#define NOFF (NDH*NC*2)       // 4096
#define MROWS (NB*NT)         // 2048
#define MT (NBH*NT)           // 16384
#define KO (NDH*NC)           // 2048

__device__ inline v8f vzero8() {
    v8f z;
#pragma unroll
    for (int i = 0; i < 8; i++) z[i] = 0.0f;
    return z;
}

__device__ inline v8f wmma_bf16(v16bf a, v16bf b, v8f c) {
    // D = A(16x32 bf16) x B(32x16 bf16) + C(16x16 f32)
    return __builtin_amdgcn_wmma_f32_16x16x32_bf16(false, a, false, b,
                                                   (short)0, c, false, false);
}

// A-operand (16x32, bf16), source row-major with leading dim ld.
// lane<16: row M=l, K = kb..kb+15; lane>=16: row M=l, K = kb+16..kb+31 (contiguous)
__device__ inline v16bf load_frag_a(const __bf16* __restrict__ base, int row0,
                                    int ld, int kb, int lane) {
    int l = lane & 15, hi = lane >> 4;
    const __bf16* p = base + (size_t)(row0 + l) * ld + kb + hi * 16;
    union { v16bf v; uint4 u[2]; } r;
    r.u[0] = *(const uint4*)(p);
    r.u[1] = *(const uint4*)(p + 8);
    return r.v;
}

// B-operand (32x16, bf16) from a (N x K) row-major "transposed" matrix:
// lane group hi holds K = kb + hi*16 .. +15 (contiguous), column N = lane&15.
__device__ inline v16bf load_frag_b(const __bf16* __restrict__ baseT, int col0,
                                    int ld, int kb, int lane) {
    int l = lane & 15, hi = lane >> 4;
    const __bf16* p = baseT + (size_t)(col0 + l) * ld + kb + hi * 16;
    union { v16bf v; uint4 u[2]; } r;
    r.u[0] = *(const uint4*)(p);
    r.u[1] = *(const uint4*)(p + 8);
    return r.v;
}

// ---------------- prep kernels ---------------------------------------------
__global__ void cvt_bf16_kernel(const float* __restrict__ in,
                                __bf16* __restrict__ out, int n) {
    int i = blockIdx.x * blockDim.x + threadIdx.x;
    if (i < n) out[i] = (__bf16)in[i];
}

// in: K x N (f32, row-major)  ->  out: N x K (bf16)
__global__ void transpose_cvt_kernel(const float* __restrict__ in,
                                     __bf16* __restrict__ out, int K, int N) {
    int i = blockIdx.x * blockDim.x + threadIdx.x;
    if (i >= K * N) return;
    int k = i / N, n = i - k * N;
    out[(size_t)n * K + k] = (__bf16)in[i];
}

// v (16 x 1024 x 256 bf16) -> vT (16 x 256 x 1024 bf16)
__global__ void transpose_v_kernel(const __bf16* __restrict__ in,
                                   __bf16* __restrict__ out) {
    int i = blockIdx.x * blockDim.x + threadIdx.x;
    if (i >= NBH * NT * NC) return;
    int c  = i & (NC - 1);
    int t  = (i >> 8) & (NT - 1);
    int bh = i >> 18;
    out[(size_t)bh * NC * NT + (size_t)c * NT + t] = in[i];
}

// ---------------- generic WMMA GEMM ----------------------------------------
// C[M,N] = act(A[M,K] @ Bt[N,K]^T + bias[N]);  EP: 0 none, 1 gelu, 2 tanh
// OUTBF: 1 -> bf16 out, 0 -> f32 out.  M % 128 == 0, N % 64 == 0, K % 32 == 0.
template <int EP, int OUTBF>
__global__ __launch_bounds__(256)
void gemm_bf16_wmma(const __bf16* __restrict__ A, const __bf16* __restrict__ Bt,
                    const float* __restrict__ bias, void* __restrict__ Cout,
                    int M, int N, int K) {
    int lane = threadIdx.x & 31;
    int wave = threadIdx.x >> 5;
    int wm = wave & 3, wn = wave >> 2;
    int row0 = blockIdx.x * 128 + wm * 32;
    int col0 = blockIdx.y * 64 + wn * 32;

    v8f acc[2][2];
#pragma unroll
    for (int i = 0; i < 2; i++)
#pragma unroll
        for (int j = 0; j < 2; j++) acc[i][j] = vzero8();

    for (int kk = 0; kk < K; kk += 32) {
        v16bf a0 = load_frag_a(A, row0,      K, kk, lane);
        v16bf a1 = load_frag_a(A, row0 + 16, K, kk, lane);
        v16bf b0 = load_frag_b(Bt, col0,      K, kk, lane);
        v16bf b1 = load_frag_b(Bt, col0 + 16, K, kk, lane);
        acc[0][0] = wmma_bf16(a0, b0, acc[0][0]);
        acc[0][1] = wmma_bf16(a0, b1, acc[0][1]);
        acc[1][0] = wmma_bf16(a1, b0, acc[1][0]);
        acc[1][1] = wmma_bf16(a1, b1, acc[1][1]);
    }

    int l = lane & 15, hi = lane >> 4;
#pragma unroll
    for (int i = 0; i < 2; i++) {
#pragma unroll
        for (int j = 0; j < 2; j++) {
            int r0 = row0 + i * 16 + hi * 8;
            int c  = col0 + j * 16 + l;
            float bv = bias[c];
#pragma unroll
            for (int r = 0; r < 8; r++) {
                float v = acc[i][j][r] + bv;
                if (EP == 1) {  // tanh-approx gelu (jax.nn.gelu default)
                    v = 0.5f * v * (1.0f + tanhf(0.7978845608028654f *
                                                 (v + 0.044715f * v * v * v)));
                } else if (EP == 2) {
                    v = tanhf(v);
                }
                size_t idx = (size_t)(r0 + r) * N + c;
                if (OUTBF) ((__bf16*)Cout)[idx] = (__bf16)v;
                else       ((float*)Cout)[idx] = v;
            }
        }
    }
}

// ---------------- deformable grid sample -----------------------------------
// off_lin: (B, T, DH*C*2) f32 (already tanh'ed); x: (B, T, C) f32
// xs_bf[(b*DH+dh)*T*C + t*C + j] = bilinear sample of tiled x at grid point.
__global__ void grid_sample_kernel(const float* __restrict__ x,
                                   const float* __restrict__ off_lin,
                                   __bf16* __restrict__ xs_bf) {
    int idx = blockIdx.x * blockDim.x + threadIdx.x;
    if (idx >= NB * NDH * NT * NC) return;
    int j  = idx & (NC - 1);
    int i  = (idx >> 8) & (NDH * NT - 1);
    int b  = idx >> 21;          // / (8192*256)
    int dh = i >> 10;
    int t  = i & (NT - 1);

    size_t obase = ((size_t)(b * NT + t)) * NOFF + (size_t)dh * NC * 2 + j * 2;
    float gx = (off_lin[obase + 0] + 1.0f) * 0.5f * (float)(NC - 1);
    float gy = (off_lin[obase + 1] + 1.0f) * 0.5f * (float)(NDH * NT - 1);
    float x0 = floorf(gx), y0 = floorf(gy);
    float wx = gx - x0, wy = gy - y0;
    int ix0 = min(max((int)x0, 0), NC - 1);
    int ix1 = min(max((int)x0 + 1, 0), NC - 1);
    int iy0 = min(max((int)y0, 0), NDH * NT - 1);
    int iy1 = min(max((int)y0 + 1, 0), NDH * NT - 1);

    const float* xb = x + (size_t)b * NT * NC;
    float v00 = xb[(size_t)(iy0 & (NT - 1)) * NC + ix0];
    float v01 = xb[(size_t)(iy0 & (NT - 1)) * NC + ix1];
    float v10 = xb[(size_t)(iy1 & (NT - 1)) * NC + ix0];
    float v11 = xb[(size_t)(iy1 & (NT - 1)) * NC + ix1];
    float v = v00 * (1.f - wy) * (1.f - wx) + v01 * (1.f - wy) * wx +
              v10 * wy * (1.f - wx) + v11 * wy * wx;
    xs_bf[idx] = (__bf16)v;
}

// ---------------- flash attention (S^T / O^T formulation) ------------------
// q,k: (BH, T, C) bf16 row-major (bias included). vT: (BH, C, T) bf16.
// Block: 128 thr (4 waves). Grid: BH*HN*16 blocks. Each wave: 16 q rows.
// Per 32-kv step: S^T(2 tiles) = K·Qᵀ (4 wmma), online softmax via lane math
// + shfl_xor(16), then Oᵀ += Vᵀ·Pᵀ (4 wmma).
__global__ __launch_bounds__(128)
void attn_kernel(const __bf16* __restrict__ q, const __bf16* __restrict__ k,
                 const __bf16* __restrict__ vT, __bf16* __restrict__ outcat) {
    int bid = blockIdx.x;
    int bh  = bid >> 6;
    int rem = bid & 63;
    int h   = rem >> 4;
    int qb  = rem & 15;
    int wave = threadIdx.x >> 5;
    int lane = threadIdx.x & 31;
    int l = lane & 15, hi = lane >> 4;
    int qr0 = qb * 64 + wave * 16;

    const __bf16* qB = q  + (size_t)bh * NT * NC;
    const __bf16* kB = k  + (size_t)bh * NT * NC;
    const __bf16* vB = vT + (size_t)bh * NC * NT;

    v16bf qf0 = load_frag_b(qB, qr0, NC, h * NHD,      lane);
    v16bf qf1 = load_frag_b(qB, qr0, NC, h * NHD + 32, lane);

    v8f o[4];
#pragma unroll
    for (int dt = 0; dt < 4; dt++) o[dt] = vzero8();
    float m = -1e30f, ls = 0.0f;

    for (int j = 0; j < NT / 32; j++) {
        int kv0 = j * 32;
        v16bf ka00 = load_frag_a(kB, kv0,      NC, h * NHD,      lane);
        v16bf ka01 = load_frag_a(kB, kv0,      NC, h * NHD + 32, lane);
        v16bf ka10 = load_frag_a(kB, kv0 + 16, NC, h * NHD,      lane);
        v16bf ka11 = load_frag_a(kB, kv0 + 16, NC, h * NHD + 32, lane);
        v8f s0 = vzero8(), s1 = vzero8();
        s0 = wmma_bf16(ka00, qf0, s0);
        s0 = wmma_bf16(ka01, qf1, s0);
        s1 = wmma_bf16(ka10, qf0, s1);
        s1 = wmma_bf16(ka11, qf1, s1);

        // --- online softmax (per q-column = per lane&15, replicated hi) ---
        float p0[8], p1[8];
        float tm = -1e30f;
#pragma unroll
        for (int r = 0; r < 8; r++) {
            p0[r] = s0[r] * 0.125f;       // 1/sqrt(64)
            p1[r] = s1[r] * 0.125f;
            tm = fmaxf(tm, fmaxf(p0[r], p1[r]));
        }
        tm = fmaxf(tm, __shfl_xor(tm, 16, 32));
        float mn = fmaxf(m, tm);
        float alpha = __expf(m - mn);
        float rs = 0.0f;
#pragma unroll
        for (int r = 0; r < 8; r++) {
            p0[r] = __expf(p0[r] - mn);
            p1[r] = __expf(p1[r] - mn);
            rs += p0[r] + p1[r];
        }
        rs += __shfl_xor(rs, 16, 32);
        ls = ls * alpha + rs;
        m = mn;
#pragma unroll
        for (int dt = 0; dt < 4; dt++)
#pragma unroll
            for (int r = 0; r < 8; r++) o[dt][r] *= alpha;

        // --- build P^T B-fragment (32kv x 16q) via half-wave swap ---------
        union { v16bf v; __bf16 e[16]; } pf;
#pragma unroll
        for (int r = 0; r < 8; r++) {
            float oth0 = __shfl_xor(p0[r], 16, 32);
            float oth1 = __shfl_xor(p1[r], 16, 32);
            float lo = hi ? oth1 : p0[r];
            float he = hi ? p1[r] : oth0;
            pf.e[r]     = (__bf16)lo;
            pf.e[8 + r] = (__bf16)he;
        }

        // --- O^T += V^T x P^T --------------------------------------------
#pragma unroll
        for (int dt = 0; dt < 4; dt++) {
            v16bf va = load_frag_a(vB, h * NHD + dt * 16, NT, kv0, lane);
            o[dt] = wmma_bf16(va, pf.v, o[dt]);
        }
    }

    float inv = 1.0f / ls;
    int b = bh >> 3, dh = bh & 7;
    int t = qr0 + l;
#pragma unroll
    for (int dt = 0; dt < 4; dt++) {
#pragma unroll
        for (int r = 0; r < 8; r++) {
            int c = dh * NC + h * NHD + dt * 16 + hi * 8 + r;
            outcat[(size_t)(b * NT + t) * KO + c] = (__bf16)(o[dt][r] * inv);
        }
    }
}

// ---------------- launch ----------------------------------------------------
extern "C" void kernel_launch(void* const* d_in, const int* in_sizes, int n_in,
                              void* d_out, int out_size, void* d_ws, size_t ws_size,
                              hipStream_t stream) {
    const float* x      = (const float*)d_in[0];
    const float* W_off1 = (const float*)d_in[1];
    const float* b_off1 = (const float*)d_in[2];
    const float* W_off2 = (const float*)d_in[3];
    const float* b_off2 = (const float*)d_in[4];
    const float* Wq = (const float*)d_in[5];  const float* bq = (const float*)d_in[6];
    const float* Wk = (const float*)d_in[7];  const float* bk = (const float*)d_in[8];
    const float* Wv = (const float*)d_in[9];  const float* bv = (const float*)d_in[10];
    const float* Wo = (const float*)d_in[11]; const float* bo = (const float*)d_in[12];

    char* w = (char*)d_ws;
    size_t off = 0;
    auto take = [&](size_t bytes) -> char* {
        char* p = w + off;
        off = (off + bytes + 255) & ~(size_t)255;
        return p;
    };
    __bf16* x_bf   = (__bf16*)take((size_t)MROWS * NC * 2);
    __bf16* W1t    = (__bf16*)take((size_t)NC * NC * 2);
    __bf16* W2t    = (__bf16*)take((size_t)NOFF * NC * 2);
    __bf16* Wqt    = (__bf16*)take((size_t)NC * NC * 2);
    __bf16* Wkt    = (__bf16*)take((size_t)NC * NC * 2);
    __bf16* Wvt    = (__bf16*)take((size_t)NC * NC * 2);
    __bf16* Wot    = (__bf16*)take((size_t)NC * KO * 2);
    __bf16* h_bf   = (__bf16*)take((size_t)MROWS * NC * 2);
    float*  off_f  = (float*) take((size_t)MROWS * NOFF * 4);   // 33.5 MB
    __bf16* xs_bf  = (__bf16*)take((size_t)MT * NC * 2);
    __bf16* q_bf   = (__bf16*)take((size_t)MT * NC * 2);
    __bf16* k_bf   = (__bf16*)take((size_t)MT * NC * 2);
    __bf16* v_bf   = (__bf16*)take((size_t)MT * NC * 2);
    __bf16* vT_bf  = (__bf16*)take((size_t)MT * NC * 2);
    __bf16* outcat = (__bf16*)take((size_t)MROWS * KO * 2);

    const int TPB = 256;
    // precision prep
    cvt_bf16_kernel<<<(MROWS * NC + TPB - 1) / TPB, TPB, 0, stream>>>(x, x_bf, MROWS * NC);
    transpose_cvt_kernel<<<(NC * NC + TPB - 1) / TPB, TPB, 0, stream>>>(W_off1, W1t, NC, NC);
    transpose_cvt_kernel<<<(NC * NOFF + TPB - 1) / TPB, TPB, 0, stream>>>(W_off2, W2t, NC, NOFF);
    transpose_cvt_kernel<<<(NC * NC + TPB - 1) / TPB, TPB, 0, stream>>>(Wq, Wqt, NC, NC);
    transpose_cvt_kernel<<<(NC * NC + TPB - 1) / TPB, TPB, 0, stream>>>(Wk, Wkt, NC, NC);
    transpose_cvt_kernel<<<(NC * NC + TPB - 1) / TPB, TPB, 0, stream>>>(Wv, Wvt, NC, NC);
    transpose_cvt_kernel<<<(KO * NC + TPB - 1) / TPB, TPB, 0, stream>>>(Wo, Wot, KO, NC);

    // offset MLP
    gemm_bf16_wmma<1, 1><<<dim3(MROWS / 128, NC / 64), 256, 0, stream>>>(
        x_bf, W1t, b_off1, h_bf, MROWS, NC, NC);
    gemm_bf16_wmma<2, 0><<<dim3(MROWS / 128, NOFF / 64), 256, 0, stream>>>(
        h_bf, W2t, b_off2, off_f, MROWS, NOFF, NC);

    // deformable resampling
    grid_sample_kernel<<<(NB * NDH * NT * NC + TPB - 1) / TPB, TPB, 0, stream>>>(
        x, off_f, xs_bf);

    // q/k/v projections
    gemm_bf16_wmma<0, 1><<<dim3(MT / 128, NC / 64), 256, 0, stream>>>(
        xs_bf, Wqt, bq, q_bf, MT, NC, NC);
    gemm_bf16_wmma<0, 1><<<dim3(MT / 128, NC / 64), 256, 0, stream>>>(
        xs_bf, Wkt, bk, k_bf, MT, NC, NC);
    gemm_bf16_wmma<0, 1><<<dim3(MT / 128, NC / 64), 256, 0, stream>>>(
        xs_bf, Wvt, bv, v_bf, MT, NC, NC);
    transpose_v_kernel<<<(MT * NC + TPB - 1) / TPB, TPB, 0, stream>>>(v_bf, vT_bf);

    // attention
    attn_kernel<<<NBH * NHN * (NT / 64), 128, 0, stream>>>(q_bf, k_bf, vT_bf, outcat);

    // output projection -> f32 result
    gemm_bf16_wmma<0, 0><<<dim3(MROWS / 128, NC / 64), 256, 0, stream>>>(
        outcat, Wot, bo, d_out, MROWS, NC, KO);
}
